// GraphGNN_29025388987167
// MI455X (gfx1250) — matrix operand order repaired and checked by reading, not compiled, not run on hardware
//
#include <hip/hip_runtime.h>
#include <hip/hip_bf16.h>
#include <math.h>

typedef __attribute__((ext_vector_type(16))) _Float16 v16h;
typedef __attribute__((ext_vector_type(8)))  float    v8f;

#define WAVES_PER_BLOCK 8
#define B_CONST 8

// fast sigmoid: v_exp_f32 + v_rcp_f32 (no IEEE div chain)
__device__ __forceinline__ float sigmoidf_(float x) {
    return __builtin_amdgcn_rcpf(1.0f + __expf(-x));
}

// ---------------- zero workspace (stats + agg) ----------------
__global__ void k_zero(float* ws, long long n) {
    long long i = (long long)blockIdx.x * blockDim.x + threadIdx.x;
    long long stride = (long long)gridDim.x * blockDim.x;
    for (; i < n; i += stride) ws[i] = 0.0f;
}

// ---------------- edge_attr column sums / sumsq ----------------
__global__ void k_stats(const float* __restrict__ ea, int E, float* __restrict__ stats) {
    __shared__ float red[4][256];
    float s0 = 0.f, s1 = 0.f, q0 = 0.f, q1 = 0.f;
    int i = blockIdx.x * blockDim.x + threadIdx.x;
    int stride = gridDim.x * blockDim.x;
    for (; i < E; i += stride) {
        float a = ea[2 * i];
        float b = ea[2 * i + 1];
        s0 += a; s1 += b; q0 += a * a; q1 += b * b;
    }
    int t = threadIdx.x;
    red[0][t] = s0; red[1][t] = s1; red[2][t] = q0; red[3][t] = q1;
    __syncthreads();
    for (int off = 128; off > 0; off >>= 1) {
        if (t < off) {
            red[0][t] += red[0][t + off];
            red[1][t] += red[1][t + off];
            red[2][t] += red[2][t + off];
            red[3][t] += red[3][t + off];
        }
        __syncthreads();
    }
    if (t == 0) {
        atomicAdd(&stats[0], red[0][0]);
        atomicAdd(&stats[1], red[1][0]);
        atomicAdd(&stats[2], red[2][0]);
        atomicAdd(&stats[3], red[3][0]);
    }
}

// ---------------- mean / rstd (ddof=1) ----------------
__global__ void k_finalize(float* stats, int E) {
    if (blockIdx.x == 0 && threadIdx.x == 0) {
        float fE = (float)E;
        float m0 = stats[0] / fE;
        float m1 = stats[1] / fE;
        float v0 = (stats[2] - fE * m0 * m0) / (fE - 1.0f);
        float v1 = (stats[3] - fE * m1 * m1) / (fE - 1.0f);
        stats[4] = m0;
        stats[5] = m1;
        stats[6] = rsqrtf(v0);
        stats[7] = rsqrtf(v1);
    }
}

// ---------------- main edge kernel: gather -> MLP (WMMA) -> scatter ----------------
// NOTE: assumes E % 16 == 0 (true for the reference shapes: E = 320000),
// so each 16-row tile lives entirely inside one batch b.
__global__ void __launch_bounds__(256)
k_edge(const float* __restrict__ x,
       const long long* __restrict__ eidx,
       const float* __restrict__ ea,
       const float* __restrict__ wmean,
       const float* __restrict__ wstd,
       const float* __restrict__ W1, const float* __restrict__ b1,
       const float* __restrict__ W2, const float* __restrict__ b2,
       const float* __restrict__ stats,
       float* __restrict__ agg,
       int N, int E)
{
    __shared__ _Float16 lds_h1[WAVES_PER_BLOCK * 256];
    const int lane = threadIdx.x & 31;
    const int wid  = threadIdx.x >> 5;
    const int row  = lane & 15;       // matrix row this lane owns (M)
    const bool hi  = lane >= 16;      // hi half-wave: K=8..15 slice of A
    const int col  = lane & 15;       // B/C/D column index for this lane
    _Float16* lds = &lds_h1[wid * 256];

    // ---- pack weights once into WMMA B-matrix layout registers ----
    // B layout (16-bit, KxN=32x16): elem j of half h -> K = 8h+j (j<8),
    // K = 16+8h+(j-8) (j>=8), N=lane%16; K beyond used range is zero.
    v16h Bw1, B2a, B2b;
#pragma unroll
    for (int j = 0; j < 16; ++j) {
        int   kk  = (j < 8 ? j : j + 8) + (hi ? 8 : 0);
        float w1  = (kk < 9) ? W1[kk * 16 + col] : 0.0f;
        float w2a = (kk < 16) ? W2[kk * 30 + col] : 0.0f;
        int   c2  = 16 + col;
        float w2b = (kk < 16 && c2 < 30) ? W2[kk * 30 + c2] : 0.0f;
        Bw1[j] = (_Float16)w1;
        B2a[j] = (_Float16)w2a;
        B2b[j] = (_Float16)w2b;
    }
    const float bias1  = b1[col];
    const float bias2a = b2[col];
    const float bias2b = (16 + col < 30) ? b2[16 + col] : 0.0f;

    const float mean0 = stats[4], mean1 = stats[5];
    const float rstd0 = stats[6], rstd1 = stats[7];
    const float wm0 = wmean[0], wm1 = wmean[1];
    const float ws0 = wstd[0],  ws1 = wstd[1];

    const unsigned tpb    = (unsigned)E >> 4;            // tiles per batch
    const unsigned nTiles = (unsigned)B_CONST * tpb;
    const unsigned gwave  = blockIdx.x * WAVES_PER_BLOCK + wid;
    const unsigned nwave  = gridDim.x * WAVES_PER_BLOCK;

    for (unsigned t = gwave; t < nTiles; t += nwave) {
        // tile-uniform batch / edge-base decomposition (one scalar-ish udiv per tile)
        const unsigned b  = t / tpb;
        const unsigned e0 = (t - b * tpb) << 4;
        const int      e  = (int)e0 + row;               // this lane's edge
        const int bN = (int)b * N;

        float f0=0.f,f1=0.f,f2=0.f,f3=0.f,f4=0.f,f5=0.f,f6=0.f,f7=0.f,f8=0.f;
        int addrT = 0, addrS = 0;
        if (!hi) {
            int s  = (int)eidx[e];
            int tg = (int)eidx[E + e];
            const float* xs = x + (long long)(bN + s)  * 3;
            const float* xt = x + (long long)(bN + tg) * 3;
            f0 = xs[0]; f1 = xs[1]; f2 = xs[2];
            f3 = xt[0]; f4 = xt[1]; f5 = xt[2];
            float cd   = ea[2 * e];
            float cdir = ea[2 * e + 1];
            float speed = f1 * ws0 + wm0;
            float direc = f2 * ws1 + wm1;
            float theta = fabsf(cdir - direc);
            float ew = 3.0f * speed * __cosf(theta) * __builtin_amdgcn_rcpf(cd);
            f8 = fmaxf(ew, 0.0f);
            f6 = (cd   - mean0) * rstd0;
            f7 = (cdir - mean1) * rstd1;
            addrT = (bN + tg) * 30;
            addrS = (bN + s)  * 30;
        }
        // hi lanes need feat[8] (K=8) of their row; low lanes hold it.
        float ew16 = __shfl(f8, row, 32);

        // ---- A1 layout (16-bit 16x32): low lanes elems0..7 = K0..7,
        //      elems8..15 = K16..23 (zero); hi lanes elem0 = K8, rest zero ----
        v16h A1;
        A1[0] = (_Float16)(hi ? ew16 : f0);
        A1[1] = (_Float16)(hi ? 0.f : f1);
        A1[2] = (_Float16)(hi ? 0.f : f2);
        A1[3] = (_Float16)(hi ? 0.f : f3);
        A1[4] = (_Float16)(hi ? 0.f : f4);
        A1[5] = (_Float16)(hi ? 0.f : f5);
        A1[6] = (_Float16)(hi ? 0.f : f6);
        A1[7] = (_Float16)(hi ? 0.f : f7);
#pragma unroll
        for (int j = 8; j < 16; ++j) A1[j] = (_Float16)0.f;

        v8f zacc = {};
        v8f h1 = __builtin_amdgcn_wmma_f32_16x16x32_f16(
            false, A1, false, Bw1, (short)0, zacc, false, false);

        // ---- bias + sigmoid, stash to per-wave LDS for the 16x16 transpose ----
#pragma unroll
        for (int r = 0; r < 8; ++r) {
            int m = r + (hi ? 8 : 0);                 // D layout: (M=m, N=col)
            float v = sigmoidf_(h1[r] + bias1);
            lds[m * 16 + col] = (_Float16)v;
        }
        asm volatile("s_wait_dscnt 0x0" ::: "memory"); // wave-private LDS RAW fence

        // ---- reload as A2 (row-major rows, K=0..15 used, K>=16 zero) ----
        const _Float16* a2p = lds + row * 16 + (hi ? 8 : 0);
        v16h A2;
#pragma unroll
        for (int j = 0; j < 8; ++j) A2[j] = a2p[j];
#pragma unroll
        for (int j = 8; j < 16; ++j) A2[j] = (_Float16)0.f;
        asm volatile("s_wait_dscnt 0x0" ::: "memory"); // keep next-iter stores behind loads

        v8f h2a = __builtin_amdgcn_wmma_f32_16x16x32_f16(
            false, A2, false, B2a, (short)0, zacc, false, false);
        v8f h2b = __builtin_amdgcn_wmma_f32_16x16x32_f16(
            false, A2, false, B2b, (short)0, zacc, false, false);

        // ---- signed scatter: +h at tgt, -h at src ----
#pragma unroll
        for (int r = 0; r < 8; ++r) {
            int m  = r + (hi ? 8 : 0);
            int at = __shfl(addrT, m, 32);
            int as = __shfl(addrS, m, 32);
            float va = sigmoidf_(h2a[r] + bias2a);
            atomicAdd(&agg[at + col], va);
            atomicAdd(&agg[as + col], -va);
            if (16 + col < 30) {
                float vb = sigmoidf_(h2b[r] + bias2b);
                atomicAdd(&agg[at + 16 + col], vb);
                atomicAdd(&agg[as + 16 + col], -vb);
            }
        }
    }
}

// ---------------- per-node 30->3 GEMV + sigmoid ----------------
__global__ void k_out(const float* __restrict__ agg, const float* __restrict__ W3,
                      const float* __restrict__ b3, float* __restrict__ out, int BN) {
    int i = blockIdx.x * blockDim.x + threadIdx.x;
    int stride = gridDim.x * blockDim.x;
    for (; i < BN; i += stride) {
        const float* a = agg + (long long)i * 30;
        float o0 = b3[0], o1 = b3[1], o2 = b3[2];
#pragma unroll
        for (int k = 0; k < 30; ++k) {
            float av = a[k];
            o0 += av * W3[k * 3 + 0];
            o1 += av * W3[k * 3 + 1];
            o2 += av * W3[k * 3 + 2];
        }
        out[(long long)i * 3 + 0] = sigmoidf_(o0);
        out[(long long)i * 3 + 1] = sigmoidf_(o1);
        out[(long long)i * 3 + 2] = sigmoidf_(o2);
    }
}

extern "C" void kernel_launch(void* const* d_in, const int* in_sizes, int n_in,
                              void* d_out, int out_size, void* d_ws, size_t ws_size,
                              hipStream_t stream) {
    const float*     x     = (const float*)d_in[0];
    const long long* eidx  = (const long long*)d_in[1];   // int64 edge_index
    const float*     ea    = (const float*)d_in[2];
    const float*     wmean = (const float*)d_in[3];
    const float*     wstd  = (const float*)d_in[4];
    const float*     W1    = (const float*)d_in[5];
    const float*     b1    = (const float*)d_in[6];
    const float*     W2    = (const float*)d_in[7];
    const float*     b2    = (const float*)d_in[8];
    const float*     W3    = (const float*)d_in[9];
    const float*     b3    = (const float*)d_in[10];
    (void)n_in; (void)out_size; (void)ws_size;

    const int B  = B_CONST;
    const int E  = in_sizes[2] / 2;          // edge_attr is (E,2)
    const int N  = in_sizes[0] / (3 * B);    // x is (B,N,3)
    const int BN = B * N;

    float* stats = (float*)d_ws;             // [0..3] sums, [4..7] mean/rstd
    float* agg   = stats + 16;               // (B*N, 30) f32
    const long long zeroN = 16 + (long long)BN * 30;

    k_zero    <<<512, 256, 0, stream>>>((float*)d_ws, zeroN);
    k_stats   <<<256, 256, 0, stream>>>(ea, E, stats);
    k_finalize<<<1,   32,  0, stream>>>(stats, E);
    k_edge    <<<640, 256, 0, stream>>>(x, eidx, ea, wmean, wstd,
                                        W1, b1, W2, b2, stats, agg, N, E);
    k_out     <<<320, 256, 0, stream>>>(agg, W3, b3, (float*)d_out, BN);
}
